// GATLayer_18056042512919
// MI455X (gfx1250) — compile-verified, matrix operand order checked by reading, not run on hardware
//
#include <hip/hip_runtime.h>

#define BATCH 16
#define NN 1024
#define DD 256

typedef __attribute__((ext_vector_type(16))) _Float16 v16h;
typedef __attribute__((ext_vector_type(8)))  float    v8f;

union V16U { uint4 u[2]; v16h h; };

// Load 16 f32 (two contiguous groups of 8 at base+off and base+off+16),
// convert to f16, in WMMA 16-bit A/B operand element order.
__device__ __forceinline__ v16h ldg_f32_as_h(const float* __restrict__ base, int off) {
  const float4* p0 = reinterpret_cast<const float4*>(base + off);
  const float4* p1 = reinterpret_cast<const float4*>(base + off + 16);
  float4 x0 = p0[0], x1 = p0[1], x2 = p1[0], x3 = p1[1];
  v16h a;
  a[0]  = (_Float16)x0.x; a[1]  = (_Float16)x0.y; a[2]  = (_Float16)x0.z; a[3]  = (_Float16)x0.w;
  a[4]  = (_Float16)x1.x; a[5]  = (_Float16)x1.y; a[6]  = (_Float16)x1.z; a[7]  = (_Float16)x1.w;
  a[8]  = (_Float16)x2.x; a[9]  = (_Float16)x2.y; a[10] = (_Float16)x2.z; a[11] = (_Float16)x2.w;
  a[12] = (_Float16)x3.x; a[13] = (_Float16)x3.y; a[14] = (_Float16)x3.z; a[15] = (_Float16)x3.w;
  return a;
}

// Load 16 f16 (two contiguous groups of 8 at base+off and base+off+16).
__device__ __forceinline__ v16h ld_h16(const _Float16* base, int off) {
  V16U t;
  t.u[0] = *reinterpret_cast<const uint4*>(base + off);
  t.u[1] = *reinterpret_cast<const uint4*>(base + off + 16);
  return t.h;
}

// ---------------------------------------------------------------------------
// Kernel 1: h = feature @ W^T + Wb ; q = h @ Q^T + Qb ; k = h @ K^T + Kb
// One wave produces 16 rows. h/q/k stored as f16 (h also transposed).
// ---------------------------------------------------------------------------
__global__ __launch_bounds__(128) void gat_proj_kernel(
    const float* __restrict__ feature,
    const float* __restrict__ Ww, const float* __restrict__ Wb,
    const float* __restrict__ Qw, const float* __restrict__ Qb,
    const float* __restrict__ Kw, const float* __restrict__ Kb,
    _Float16* __restrict__ h16, _Float16* __restrict__ hT16,
    _Float16* __restrict__ q16, _Float16* __restrict__ k16)
{
  __shared__ alignas(16) _Float16 hsh[4][16][DD];

  const int lane = threadIdx.x & 31;
  const int wv   = threadIdx.x >> 5;
  const int gw   = blockIdx.x * 4 + wv;     // 0..1023 : (b, row-tile)
  const int b    = gw >> 6;
  const int r0   = (gw & 63) << 4;
  const int hi   = (lane >> 4) & 1;
  const int ln   = lane & 15;

  const float* frow = feature + (size_t)(b * NN + r0 + ln) * DD;

  // ---- Phase A: h tile (16 x 256) ----
  for (int t = 0; t < 16; ++t) {
    const int ocol = t * 16 + ln;
    const float bias = Wb[ocol];
    v8f c;
    #pragma unroll
    for (int i = 0; i < 8; ++i) c[i] = bias;
    const float* wr = Ww + (size_t)ocol * DD;   // row of W: W[o, :]
    #pragma unroll
    for (int kk = 0; kk < 8; ++kk) {
      const int off = kk * 32 + hi * 8;
      v16h a  = ldg_f32_as_h(frow, off);
      v16h bm = ldg_f32_as_h(wr, off);
      c = __builtin_amdgcn_wmma_f32_16x16x32_f16(false, a, false, bm, (short)0, c, false, false);
    }
    #pragma unroll
    for (int r = 0; r < 8; ++r) {
      const int row = r0 + r + hi * 8;
      const _Float16 v = (_Float16)c[r];
      h16[(size_t)(b * NN + row) * DD + ocol]  = v;
      hT16[(size_t)(b * DD + ocol) * NN + row] = v;
      hsh[wv][r + hi * 8][ocol] = v;
    }
  }
  asm volatile("s_wait_dscnt 0" ::: "memory");   // cross-lane LDS handoff

  // ---- Phase B: q and k from h ----
  const _Float16* hrow = &hsh[wv][ln][0];
  for (int t = 0; t < 16; ++t) {
    const int ocol = t * 16 + ln;
    const float bq = Qb[ocol], bk = Kb[ocol];
    v8f cq, ck;
    #pragma unroll
    for (int i = 0; i < 8; ++i) { cq[i] = bq; ck[i] = bk; }
    const float* qr = Qw + (size_t)ocol * DD;
    const float* kr = Kw + (size_t)ocol * DD;
    #pragma unroll
    for (int kk = 0; kk < 8; ++kk) {
      const int off = kk * 32 + hi * 8;
      v16h a   = ld_h16(hrow, off);
      v16h bqm = ldg_f32_as_h(qr, off);
      v16h bkm = ldg_f32_as_h(kr, off);
      cq = __builtin_amdgcn_wmma_f32_16x16x32_f16(false, a, false, bqm, (short)0, cq, false, false);
      ck = __builtin_amdgcn_wmma_f32_16x16x32_f16(false, a, false, bkm, (short)0, ck, false, false);
    }
    #pragma unroll
    for (int r = 0; r < 8; ++r) {
      const int row = r0 + r + hi * 8;
      q16[(size_t)(b * NN + row) * DD + ocol] = (_Float16)cq[r];
      k16[(size_t)(b * NN + row) * DD + ocol] = (_Float16)ck[r];
    }
  }
}

// ---------------------------------------------------------------------------
// Kernel 2: flash-attention style fused masked softmax + a@h, relu.
// One wave handles 16 query rows, streams keys in chunks of 32.
// ---------------------------------------------------------------------------
__global__ __launch_bounds__(128) void gat_attn_kernel(
    const int* __restrict__ graph,
    const _Float16* __restrict__ hT16,
    const _Float16* __restrict__ q16,
    const _Float16* __restrict__ k16,
    float* __restrict__ out)
{
  __shared__ alignas(16) _Float16 qsh[4][16][DD];
  __shared__ alignas(16) _Float16 psh[4][16][32];

  const int lane = threadIdx.x & 31;
  const int wv   = threadIdx.x >> 5;
  const int gw   = blockIdx.x * 4 + wv;
  const int b    = gw >> 6;
  const int r0   = (gw & 63) << 4;
  const int hi   = (lane >> 4) & 1;
  const int ln   = lane & 15;

  // Stage this wave's 16 q rows into LDS (16*256 halves = 512 x uint4).
  const _Float16* qg = q16 + (size_t)(b * NN + r0) * DD;
  #pragma unroll
  for (int it = 0; it < 16; ++it) {
    const int chunk = it * 32 + lane;
    const int row = chunk >> 5;
    const int off = (chunk & 31) * 8;
    *reinterpret_cast<uint4*>(&qsh[wv][row][off]) =
        *reinterpret_cast<const uint4*>(qg + (size_t)row * DD + off);
  }
  asm volatile("s_wait_loadcnt 0\n\ts_wait_dscnt 0" ::: "memory");

  v8f acc[16];
  #pragma unroll
  for (int t = 0; t < 16; ++t) {
    #pragma unroll
    for (int i = 0; i < 8; ++i) acc[t][i] = 0.0f;
  }
  float mi[8], li[8], corr[8];
  #pragma unroll
  for (int r = 0; r < 8; ++r) { mi[r] = -3.0e38f; li[r] = 0.0f; }

  const _Float16* qrow = &qsh[wv][ln][0];
  const _Float16* prow = &psh[wv][ln][0];

  #pragma unroll 1
  for (int mt = 0; mt < 32; ++mt) {
    const int m0 = mt * 32;
    v8f s0, s1;
    #pragma unroll
    for (int i = 0; i < 8; ++i) { s0[i] = 0.0f; s1[i] = 0.0f; }

    const _Float16* k0 = k16 + (size_t)(b * NN + m0 + ln) * DD;
    const _Float16* k1 = k0 + (size_t)16 * DD;
    #pragma unroll
    for (int kk = 0; kk < 8; ++kk) {
      const int off = kk * 32 + hi * 8;
      v16h a  = ld_h16(qrow, off);
      v16h b0 = ld_h16(k0, off);
      v16h b1 = ld_h16(k1, off);
      s0 = __builtin_amdgcn_wmma_f32_16x16x32_f16(false, a, false, b0, (short)0, s0, false, false);
      s1 = __builtin_amdgcn_wmma_f32_16x16x32_f16(false, a, false, b1, (short)0, s1, false, false);
    }

    // scale, leaky-relu, mask, online softmax stats per row (C layout:
    // row = r + hi*8, column = lane%16 within the 16-lane group).
    #pragma unroll
    for (int r = 0; r < 8; ++r) {
      const size_t grow = (size_t)(b * NN + r0 + r + hi * 8) * NN + m0;
      const int g0 = graph[grow + ln];
      const int g1 = graph[grow + 16 + ln];
      float e0 = s0[r] * 0.0625f;            // 1/sqrt(256)
      float e1 = s1[r] * 0.0625f;
      e0 = (e0 >= 0.0f) ? e0 : 0.2f * e0;
      e1 = (e1 >= 0.0f) ? e1 : 0.2f * e1;
      e0 = (g0 == 0) ? -1.0e9f : e0;
      e1 = (g1 == 0) ? -1.0e9f : e1;
      float v = fmaxf(e0, e1);
      v = fmaxf(v, __shfl_xor(v, 1, 32));
      v = fmaxf(v, __shfl_xor(v, 2, 32));
      v = fmaxf(v, __shfl_xor(v, 4, 32));
      v = fmaxf(v, __shfl_xor(v, 8, 32));
      const float nm = fmaxf(mi[r], v);
      const float cr = __expf(mi[r] - nm);
      const float x0 = __expf(e0 - nm);
      const float x1 = __expf(e1 - nm);
      float sum = x0 + x1;
      sum += __shfl_xor(sum, 1, 32);
      sum += __shfl_xor(sum, 2, 32);
      sum += __shfl_xor(sum, 4, 32);
      sum += __shfl_xor(sum, 8, 32);
      li[r] = li[r] * cr + sum;
      mi[r] = nm;
      corr[r] = cr;
      psh[wv][r + hi * 8][ln]      = (_Float16)x0;   // re-layout p as A operand
      psh[wv][r + hi * 8][ln + 16] = (_Float16)x1;
    }
    asm volatile("s_wait_dscnt 0" ::: "memory");

    // rescale accumulator by row correction
    #pragma unroll
    for (int t = 0; t < 16; ++t) {
      #pragma unroll
      for (int r = 0; r < 8; ++r) acc[t][r] *= corr[r];
    }

    // acc += p (16x32) @ h (32x256), B slices from transposed h.
    v16h pa = ld_h16(prow, hi * 8);
    #pragma unroll
    for (int t = 0; t < 16; ++t) {
      const _Float16* hcol = hT16 + (size_t)(b * DD + t * 16 + ln) * NN + m0;
      v16h bh = ld_h16(hcol, hi * 8);
      acc[t] = __builtin_amdgcn_wmma_f32_16x16x32_f16(false, pa, false, bh, (short)0, acc[t], false, false);
    }
  }

  // epilogue: divide by softmax denominator, relu, store f32
  #pragma unroll
  for (int r = 0; r < 8; ++r) li[r] = 1.0f / li[r];
  #pragma unroll
  for (int t = 0; t < 16; ++t) {
    #pragma unroll
    for (int r = 0; r < 8; ++r) {
      float v = acc[t][r] * li[r];
      v = (v > 0.0f) ? v : 0.0f;
      out[(size_t)(b * NN + r0 + r + hi * 8) * DD + t * 16 + ln] = v;
    }
  }
}

extern "C" void kernel_launch(void* const* d_in, const int* in_sizes, int n_in,
                              void* d_out, int out_size, void* d_ws, size_t ws_size,
                              hipStream_t stream) {
  const float* feature = (const float*)d_in[0];
  const int*   graph   = (const int*)d_in[1];
  const float* Ww      = (const float*)d_in[2];
  const float* Wb      = (const float*)d_in[3];
  const float* Qw      = (const float*)d_in[4];
  const float* Qb      = (const float*)d_in[5];
  const float* Kw      = (const float*)d_in[6];
  const float* Kb      = (const float*)d_in[7];
  float* out = (float*)d_out;

  const size_t tile = (size_t)BATCH * NN * DD;   // 4M elements
  _Float16* h16  = (_Float16*)d_ws;
  _Float16* hT16 = h16  + tile;
  _Float16* q16  = hT16 + tile;
  _Float16* k16  = q16  + tile;                  // total 32 MB of workspace

  // 1024 waves total, 4 waves (128 threads) per block.
  gat_proj_kernel<<<256, 128, 0, stream>>>(feature, Ww, Wb, Qw, Qb, Kw, Kb,
                                           h16, hT16, q16, k16);
  gat_attn_kernel<<<256, 128, 0, stream>>>(graph, hT16, q16, k16, out);
}
